// InfoStillAttention_84516366451097
// MI455X (gfx1250) — matrix operand order repaired and checked by reading, not compiled, not run on hardware
//
#include <hip/hip_runtime.h>
#include <hip/hip_bf16.h>
#include <float.h>

// B=2, H=16, S=2048, D=64 ; outputs: out [BH,S,D] fp32 then raw [BH,S,S] fp32
#define BH  32
#define SEQ 2048
#define DH  64
#define QT  64      // q rows per block
#define KT  64      // keys per iteration
#define NW  4       // waves per block (128 threads), 16 q-rows per wave

typedef __attribute__((ext_vector_type(16))) __bf16 bf16x16;
typedef __attribute__((ext_vector_type(8)))  float  v8f;

union Frag { bf16x16 v; uint4 u[2]; };

__device__ __forceinline__ unsigned short f2bf(float f) {
  unsigned u = __float_as_uint(f);
  u += 0x7FFFu + ((u >> 16) & 1u);          // round-to-nearest-even
  return (unsigned short)(u >> 16);
}

// ---------------- prep: L2-normalize q,k -> bf16 ; cast v -> bf16 transposed ----
__global__ void __launch_bounds__(256)
prep_kernel(const float* __restrict__ q, const float* __restrict__ k,
            const float* __restrict__ v,
            unsigned short* __restrict__ qn, unsigned short* __restrict__ kn,
            unsigned short* __restrict__ vt) {
  const int w    = threadIdx.x >> 5;
  const int lane = threadIdx.x & 31;
  const int row  = blockIdx.x * 8 + w;       // 0 .. 3*BH*SEQ-1
  const int arr  = row >> 16;                // 65536 rows per tensor
  const int r    = row & 0xFFFF;
  const float* src = (arr == 0 ? q : (arr == 1 ? k : v)) + (size_t)r * DH;
  float2 x = ((const float2*)src)[lane];
  if (arr < 2) {
    float ss = x.x * x.x + x.y * x.y;
    #pragma unroll
    for (int off = 16; off >= 1; off >>= 1) ss += __shfl_xor(ss, off, 32);
    float inv = 1.0f / fmaxf(sqrtf(ss), 1e-12f);
    unsigned p = (unsigned)f2bf(x.x * inv) | ((unsigned)f2bf(x.y * inv) << 16);
    unsigned short* dst = (arr == 0 ? qn : kn) + (size_t)r * DH;
    ((unsigned*)dst)[lane] = p;
  } else {
    const int bh = r >> 11, s = r & (SEQ - 1);
    unsigned short* base = vt + (size_t)bh * DH * SEQ + s;   // vt[bh][d][s]
    base[(size_t)(2 * lane)     * SEQ] = f2bf(x.x);
    base[(size_t)(2 * lane + 1) * SEQ] = f2bf(x.y);
  }
}

// ---------------- flash attention + raw-score streaming --------------------
__global__ void __launch_bounds__(128)
attn_kernel(const unsigned short* __restrict__ qn,
            const unsigned short* __restrict__ kn,
            const unsigned short* __restrict__ vt,
            const float* __restrict__ qks,
            float* __restrict__ out, float* __restrict__ raw) {
  __shared__ unsigned short pstage[NW * 16 * KT];   // 8 KB, per-wave private slices

  const int qt   = blockIdx.x;
  const int bh   = blockIdx.y;
  const int w    = threadIdx.x >> 5;
  const int lane = threadIdx.x & 31;
  const int ln   = lane & 15;
  const int half = lane >> 4;
  const int q0   = qt * QT;
  const float scale = qks[0];
  // |qn.kn| <= 1 (cosine), so scaled scores are bounded by |scale|:
  // a constant softmax shift is exact -> no online max / rescale needed.
  const float cmax = fabsf(scale);
  const v8f vzero = {};

  // Q A-fragments (16x32 bf16 x2): lane<16 holds K {0..7,16..23}, lane>=16 {8..15,24..31}
  Frag qf[2];
  {
    const unsigned short* qrow = qn + ((size_t)bh * SEQ + q0 + w * 16 + ln) * DH;
    #pragma unroll
    for (int f = 0; f < 2; ++f) {
      qf[f].u[0] = *(const uint4*)(qrow + f * 32 + half * 8);
      qf[f].u[1] = *(const uint4*)(qrow + f * 32 + 16 + half * 8);
    }
  }

  // all-ones bf16 B fragment: P x ones gives row sums replicated in C layout
  Frag ones;
  ones.u[0] = make_uint4(0x3F803F80u, 0x3F803F80u, 0x3F803F80u, 0x3F803F80u);
  ones.u[1] = ones.u[0];

  v8f oacc[4];
  v8f lacc = vzero;                 // lacc[j] = running softmax denominator of row j
  #pragma unroll
  for (int a = 0; a < 4; ++a) oacc[a] = vzero;

  unsigned short* pbuf = pstage + w * (16 * KT);
  const int rowg = q0 + w * 16 + 8 * half;       // + j gives global q row

  for (int kt = 0; kt < SEQ / KT; ++kt) {
    const int k0 = kt * KT;
    const unsigned short* kbase = kn + ((size_t)bh * SEQ + k0) * DH;

    if (kt + 1 < SEQ / KT)   // pull next K tile toward the near caches
      __builtin_prefetch(kbase + (size_t)KT * DH + (size_t)lane * (KT * DH / 32), 0, 3);

    // ---- issue ALL fragment loads for this tile first (K and V) ----
    Frag kf[4][2];
    #pragma unroll
    for (int a = 0; a < 4; ++a) {
      const unsigned short* krow = kbase + (size_t)(a * 16 + ln) * DH;
      #pragma unroll
      for (int f = 0; f < 2; ++f) {
        kf[a][f].u[0] = *(const uint4*)(krow + f * 32 + half * 8);
        kf[a][f].u[1] = *(const uint4*)(krow + f * 32 + 16 + half * 8);
      }
    }
    Frag vf[4][2];
    const bool live = (kt <= qt);
    if (live) {
      const unsigned short* vbase = vt + (size_t)bh * DH * SEQ + k0;
      #pragma unroll
      for (int a = 0; a < 4; ++a) {
        const unsigned short* vrow = vbase + (size_t)(a * 16 + ln) * SEQ;
        #pragma unroll
        for (int s = 0; s < 2; ++s) {
          vf[a][s].u[0] = *(const uint4*)(vrow + s * 32 + half * 8);
          vf[a][s].u[1] = *(const uint4*)(vrow + s * 32 + 16 + half * 8);
        }
      }
      if (kt + 1 <= qt) {    // prefetch next V tile (2 rows per lane covered)
        __builtin_prefetch(vbase + (size_t)(2 * lane) * SEQ + KT, 0, 3);
        __builtin_prefetch(vbase + (size_t)(2 * lane + 1) * SEQ + KT, 0, 3);
      }
    }
    // keep every load above this point: gives ~32 outstanding requests that the
    // WMMA chain below drains with partial loadcnt waits instead of one-by-one.
    __builtin_amdgcn_sched_barrier(0);

    // ---- S-tile (16x64 per wave) = Qn x Kn^T, f32 accum ----
    v8f sacc[4];
    #pragma unroll
    for (int a = 0; a < 4; ++a) {
      sacc[a] = vzero;
      #pragma unroll
      for (int f = 0; f < 2; ++f)
        sacc[a] = __builtin_amdgcn_wmma_f32_16x16x32_bf16(
            false, qf[f].v, false, kf[a][f].v, (short)0, sacc[a], false, false);
    }

    // ---- stream raw (unscaled, unmasked) scores to HBM ----
    // write-once data: non-temporal stores keep the 536 MB stream from
    // evicting the K/V working set out of WGP$/L2.
    {
      float* rp = raw + ((size_t)bh * SEQ + q0 + w * 16 + 8 * half) * SEQ + k0 + ln;
      #pragma unroll
      for (int a = 0; a < 4; ++a)
        #pragma unroll
        for (int j = 0; j < 8; ++j)
          __builtin_nontemporal_store(sacc[a][j], rp + (size_t)j * SEQ + a * 16);
    }

    // ---- causal softmax (constant shift) + PV, tiles at/below diagonal ----
    if (live) {
      // p = exp(s*scale - cmax) masked; straight into LDS as bf16 (C->A relayout)
      #pragma unroll
      for (int a = 0; a < 4; ++a) {
        const int col = k0 + a * 16 + ln;
        #pragma unroll
        for (int j = 0; j < 8; ++j) {
          float e = (col <= rowg + j) ? __expf(fmaf(sacc[a][j], scale, -cmax)) : 0.0f;
          pbuf[(j + 8 * half) * KT + a * 16 + ln] = f2bf(e);
        }
      }
      // wave-private LDS slice; LDS is in-order within a wave -> no workgroup
      // barrier, just stop the compiler reordering and drain DS stores.
      __builtin_amdgcn_wave_barrier();
      asm volatile("s_wait_dscnt 0x0" ::: "memory");

      Frag pf[2];
      #pragma unroll
      for (int s = 0; s < 2; ++s) {
        const unsigned short* prow = pbuf + ln * KT + s * 32;
        pf[s].u[0] = *(const uint4*)(prow + half * 8);
        pf[s].u[1] = *(const uint4*)(prow + 16 + half * 8);
      }
      __builtin_amdgcn_wave_barrier();

      // row sums via WMMA: lacc += P x ones  (C layout matches oacc rows)
      #pragma unroll
      for (int s = 0; s < 2; ++s)
        lacc = __builtin_amdgcn_wmma_f32_16x16x32_bf16(
            false, pf[s].v, false, ones.v, (short)0, lacc, false, false);

      // O += P x V   (V fragments were loaded before the QK chain)
      #pragma unroll
      for (int a = 0; a < 4; ++a)
        #pragma unroll
        for (int s = 0; s < 2; ++s)
          oacc[a] = __builtin_amdgcn_wmma_f32_16x16x32_bf16(
              false, pf[s].v, false, vf[a][s].v, (short)0, oacc[a], false, false);
    }
  }

  // ---- epilogue: out = O / l ----
  float invl[8];
  #pragma unroll
  for (int j = 0; j < 8; ++j) invl[j] = 1.0f / lacc[j];
  float* op = out + ((size_t)bh * SEQ + q0 + w * 16 + 8 * half) * DH + ln;
  #pragma unroll
  for (int a = 0; a < 4; ++a)
    #pragma unroll
    for (int j = 0; j < 8; ++j)
      op[(size_t)j * DH + a * 16] = oacc[a][j] * invl[j];
}

extern "C" void kernel_launch(void* const* d_in, const int* in_sizes, int n_in,
                              void* d_out, int out_size, void* d_ws, size_t ws_size,
                              hipStream_t stream) {
  const float* q   = (const float*)d_in[0];
  const float* k   = (const float*)d_in[1];
  const float* v   = (const float*)d_in[2];
  const float* qks = (const float*)d_in[3];

  float* out = (float*)d_out;
  float* raw = out + (size_t)BH * SEQ * DH;      // outputs concatenated: out, raw

  unsigned short* qn = (unsigned short*)d_ws;            // 8 MB
  unsigned short* kn = qn + (size_t)BH * SEQ * DH;       // 8 MB
  unsigned short* vt = kn + (size_t)BH * SEQ * DH;       // 8 MB (transposed)

  prep_kernel<<<3 * BH * SEQ / 8, 256, 0, stream>>>(q, k, v, qn, kn, vt);
  attn_kernel<<<dim3(SEQ / QT, BH), 128, 0, stream>>>(qn, kn, vt, qks, out, raw);
}
